// GAT_41154376630502
// MI455X (gfx1250) — compile-verified, matrix operand order checked by reading, not run on hardware
//
#include <hip/hip_runtime.h>
#include <hip/hip_bf16.h>

typedef __attribute__((ext_vector_type(16))) _Float16 v16h;
typedef __attribute__((ext_vector_type(8)))  float    v8f;

#define LRELU_NEG 0.2f

__device__ __forceinline__ float leaky(float a) { return a > 0.f ? a : LRELU_NEG * a; }
// monotone bijection float -> uint so atomicMax(uint) == float max
__device__ __forceinline__ unsigned fkey(float f) {
    unsigned u = __float_as_uint(f);
    return (u & 0x80000000u) ? ~u : (u | 0x80000000u);
}
__device__ __forceinline__ float funkey(unsigned u) {
    return (u & 0x80000000u) ? __uint_as_float(u & 0x7fffffffu) : __uint_as_float(~u);
}

// ---------------- f32 -> f16 convert ----------------
__global__ void cvt_f16_kernel(const float* __restrict__ src, _Float16* __restrict__ dst, int n) {
    int i = blockIdx.x * blockDim.x + threadIdx.x;
    if (i < n) dst[i] = (_Float16)src[i];
}

// ---------------- WMMA GEMM: D[M,Ncols] = A[M,K] * B[Ncols,K]^T ----------------
// one wave per 16x16 output tile; blockDim = (32, WV) waves covering WV col-tiles
__global__ void gemm_wmma_f16(const _Float16* __restrict__ A, const _Float16* __restrict__ B,
                              float* __restrict__ D, int K, int Ncols) {
    const int m0   = blockIdx.x * 16;
    const int n0   = (blockIdx.y * blockDim.y + threadIdx.y) * 16;
    const int lane = threadIdx.x;
    const int g = lane >> 4;      // K-half select
    const int r = lane & 15;      // row (A) / col (B) within tile
    const _Float16* arow = A + (size_t)(m0 + r) * K;
    const _Float16* brow = B + (size_t)(n0 + r) * K;
    v8f c = {};
    for (int k0 = 0; k0 < K; k0 += 32) {
        union { v16h v; unsigned u[8]; } a, b;
#pragma unroll
        for (int i = 0; i < 8; ++i) {
            // 16-bit operand layout: VGPR i<4 -> K = g*8+2i, i>=4 -> K = 16+g*8+2(i-4); pairs packed
            int k = k0 + ((i < 4) ? (g * 8 + 2 * i) : (16 + g * 8 + 2 * (i - 4)));
            a.u[i] = *(const unsigned*)(arow + k);
            b.u[i] = *(const unsigned*)(brow + k);
        }
        c = __builtin_amdgcn_wmma_f32_16x16x32_f16(false, a.v, false, b.v,
                                                   (short)0, c, false, false);
    }
#pragma unroll
    for (int t = 0; t < 8; ++t)   // D layout: VGPR t -> row m0 + g*8 + t, col n0 + r
        D[(size_t)(m0 + g * 8 + t) * Ncols + (n0 + r)] = c[t];
}

// ---------------- attention logits: ai/aj[n,h] = sum_c xh[n,h,c]*att[h,c] ----------------
__global__ void logits_kernel(const float* __restrict__ xh, const float* __restrict__ att_i,
                              const float* __restrict__ att_j, float* __restrict__ ai,
                              float* __restrict__ aj, int NH, int H, int C) {
    int i = blockIdx.x * blockDim.x + threadIdx.x;
    if (i >= NH) return;
    int h = i % H;
    const float* row = xh + (size_t)i * C;   // (n*H + h)*C == n*H*C + h*C
    float si = 0.f, sj = 0.f;
    for (int c = 0; c < C; ++c) { si += row[c] * att_i[h * C + c]; sj += row[c] * att_j[h * C + c]; }
    ai[i] = si; aj[i] = sj;
}

// self-loop logit + init running max
__global__ void init_self_kernel(const float* __restrict__ ai, const float* __restrict__ aj,
                                 float* __restrict__ aself, unsigned* __restrict__ m, int NH) {
    int i = blockIdx.x * blockDim.x + threadIdx.x;
    if (i >= NH) return;
    float a = leaky(ai[i] + aj[i]);
    aself[i] = a;
    m[i] = fkey(a);
}

__global__ void edge_max_kernel(const int* __restrict__ src, const int* __restrict__ dst,
                                const float* __restrict__ ai, const float* __restrict__ aj,
                                unsigned* __restrict__ m, int EH, int H) {
    int i = blockIdx.x * blockDim.x + threadIdx.x;
    if (i >= EH) return;
    int e = i / H, h = i - e * H;
    float a = leaky(ai[dst[e] * H + h] + aj[src[e] * H + h]);
    atomicMax(&m[dst[e] * H + h], fkey(a));
}

// convert key back to float max (in place) and seed denom with self-loop term
__global__ void denom_init_kernel(const float* __restrict__ aself, unsigned* __restrict__ m,
                                  float* __restrict__ denom, int NH) {
    int i = blockIdx.x * blockDim.x + threadIdx.x;
    if (i >= NH) return;
    float mf = funkey(m[i]);
    ((float*)m)[i] = mf;
    denom[i] = __expf(aself[i] - mf);
}

__global__ void edge_denom_kernel(const int* __restrict__ src, const int* __restrict__ dst,
                                  const float* __restrict__ ai, const float* __restrict__ aj,
                                  const float* __restrict__ m, float* __restrict__ denom,
                                  int EH, int H) {
    int i = blockIdx.x * blockDim.x + threadIdx.x;
    if (i >= EH) return;
    int e = i / H, h = i - e * H;
    int d = dst[e];
    float a = leaky(ai[d * H + h] + aj[src[e] * H + h]);
    atomicAdd(&denom[d * H + h], __expf(a - m[d * H + h]));
}

__global__ void edge_alpha_kernel(const int* __restrict__ src, const int* __restrict__ dst,
                                  const float* __restrict__ ai, const float* __restrict__ aj,
                                  const float* __restrict__ m, const float* __restrict__ denom,
                                  float* __restrict__ alpha, int EH, int H) {
    int i = blockIdx.x * blockDim.x + threadIdx.x;
    if (i >= EH) return;
    int e = i / H, h = i - e * H;
    int d = dst[e];
    float a = leaky(ai[d * H + h] + aj[src[e] * H + h]);
    alpha[i] = __expf(a - m[d * H + h]) / (denom[d * H + h] + 1e-16f);
}

// out[n,:] = w_self(n,h) * xh[n,:]  (initializes the accumulator); float4 per thread
__global__ void self_agg_kernel(const float4* __restrict__ xh4, const float* __restrict__ aself,
                                const float* __restrict__ m, const float* __restrict__ denom,
                                float4* __restrict__ out4, int NHC4, int C4) {
    int i = blockIdx.x * blockDim.x + threadIdx.x;
    if (i >= NHC4) return;
    int nh = i / C4;
    float w = __expf(aself[nh] - m[nh]) / (denom[nh] + 1e-16f);
    float4 v = xh4[i];
    out4[i] = make_float4(w * v.x, w * v.y, w * v.z, w * v.w);
}

// out[dst,:] += alpha[e,h] * xh[src,:]; one float4 gather + 4 atomics per thread
__global__ void edge_agg_kernel(const int* __restrict__ src, const int* __restrict__ dst,
                                const float4* __restrict__ xh4, const float* __restrict__ alpha,
                                float* __restrict__ out, int EHC4, int H, int C4) {
    int i = blockIdx.x * blockDim.x + threadIdx.x;
    if (i >= EHC4) return;
    int HC4 = H * C4;
    int e = i / HC4, j4 = i - e * HC4;          // j4 in [0, HC/4)
    int h = j4 / C4;
    int s = src[e], d = dst[e];
    float w = alpha[e * H + h];
    float4 v = xh4[(size_t)s * HC4 + j4];       // global_load_b128 gather
    float* o = out + ((size_t)d * HC4 + j4) * 4;
    atomicAdd(o + 0, w * v.x);
    atomicAdd(o + 1, w * v.y);
    atomicAdd(o + 2, w * v.z);
    atomicAdd(o + 3, w * v.w);
}

// layer1 epilogue: bias + ELU, store f32 and f16 (float4 / 4xhalf per thread)
__global__ void post1_kernel(float4* __restrict__ h4, const float* __restrict__ bias,
                             _Float16* __restrict__ h16, int NHC4, int HC4) {
    int i = blockIdx.x * blockDim.x + threadIdx.x;
    if (i >= NHC4) return;
    int b = (i % HC4) * 4;
    float4 v = h4[i];
    float r[4] = {v.x + bias[b], v.y + bias[b + 1], v.z + bias[b + 2], v.w + bias[b + 3]};
#pragma unroll
    for (int t = 0; t < 4; ++t) r[t] = r[t] > 0.f ? r[t] : (__expf(r[t]) - 1.0f);
    h4[i] = make_float4(r[0], r[1], r[2], r[3]);
    _Float16* p = h16 + (size_t)i * 4;
    p[0] = (_Float16)r[0]; p[1] = (_Float16)r[1]; p[2] = (_Float16)r[2]; p[3] = (_Float16)r[3];
}

// layer2 epilogue: bias in place (H2=1 -> head-mean is identity)
__global__ void post2_kernel(float4* __restrict__ out4, const float* __restrict__ bias,
                             int NHC4, int HC4) {
    int i = blockIdx.x * blockDim.x + threadIdx.x;
    if (i >= NHC4) return;
    int b = (i % HC4) * 4;
    float4 v = out4[i];
    out4[i] = make_float4(v.x + bias[b], v.y + bias[b + 1], v.z + bias[b + 2], v.w + bias[b + 3]);
}

static inline int cdiv(long long a, int b) { return (int)((a + b - 1) / b); }

extern "C" void kernel_launch(void* const* d_in, const int* in_sizes, int n_in,
                              void* d_out, int out_size, void* d_ws, size_t ws_size,
                              hipStream_t stream) {
    const int F_IN = 128, H1 = 8, C1 = 32, HC1 = H1 * C1;   // 256
    const int H2 = 1, C2 = 128, HC2 = H2 * C2;              // 128
    const int N = in_sizes[0] / F_IN;          // 50000 (multiple of 16)
    const int E = in_sizes[1] / 2;             // 800000

    const float* x      = (const float*)d_in[0];
    const int*   src    = (const int*)d_in[1];
    const int*   dst    = (const int*)d_in[1] + E;
    const float* lin1_w = (const float*)d_in[2];
    const float* att1_i = (const float*)d_in[3];
    const float* att1_j = (const float*)d_in[4];
    const float* bias1  = (const float*)d_in[5];
    const float* lin2_w = (const float*)d_in[6];
    const float* att2_i = (const float*)d_in[7];
    const float* att2_j = (const float*)d_in[8];
    const float* bias2  = (const float*)d_in[9];
    float* out = (float*)d_out;

    // ---- workspace carve-up (256B aligned) ----
    char* ws = (char*)d_ws;
    size_t off = 0;
    auto take = [&](size_t bytes) -> void* {
        void* p = ws + off;
        off = (off + bytes + 255) & ~(size_t)255;
        return p;
    };
    _Float16* feat16 = (_Float16*)take((size_t)N * HC1 * 2);   // x_f16 (N*128) then h_f16 (N*256)
    _Float16* w16    = (_Float16*)take((size_t)HC1 * F_IN * 2 * 2); // fits either weight
    float*    xh     = (float*)take((size_t)N * HC1 * 4);      // projected features
    float*    hbuf   = (float*)take((size_t)N * HC1 * 4);      // layer1 output / ELU'd hidden
    float*    ai     = (float*)take((size_t)N * H1 * 4);
    float*    aj     = (float*)take((size_t)N * H1 * 4);
    float*    aself  = (float*)take((size_t)N * H1 * 4);
    unsigned* mbuf   = (unsigned*)take((size_t)N * H1 * 4);    // fkey max, then float max
    float*    denom  = (float*)take((size_t)N * H1 * 4);
    float*    alpha  = (float*)take((size_t)E * H1 * 4);

    const int TPB = 256;
    // =========================== layer 1 ===========================
    cvt_f16_kernel<<<cdiv((long long)N * F_IN, TPB), TPB, 0, stream>>>(x, feat16, N * F_IN);
    cvt_f16_kernel<<<cdiv(HC1 * F_IN, TPB), TPB, 0, stream>>>(lin1_w, w16, HC1 * F_IN);

    {   // xh = x @ lin1_w^T : M=N, K=128, Ncols=256; 4 waves/block over col tiles
        dim3 blk(32, 4), grd(N / 16, HC1 / 64);
        gemm_wmma_f16<<<grd, blk, 0, stream>>>(feat16, w16, xh, F_IN, HC1);
    }
    int NH = N * H1, EH = E * H1;
    logits_kernel<<<cdiv(NH, TPB), TPB, 0, stream>>>(xh, att1_i, att1_j, ai, aj, NH, H1, C1);
    init_self_kernel<<<cdiv(NH, TPB), TPB, 0, stream>>>(ai, aj, aself, mbuf, NH);
    edge_max_kernel<<<cdiv(EH, TPB), TPB, 0, stream>>>(src, dst, ai, aj, mbuf, EH, H1);
    denom_init_kernel<<<cdiv(NH, TPB), TPB, 0, stream>>>(aself, mbuf, denom, NH);
    edge_denom_kernel<<<cdiv(EH, TPB), TPB, 0, stream>>>(src, dst, ai, aj, (float*)mbuf, denom, EH, H1);
    edge_alpha_kernel<<<cdiv(EH, TPB), TPB, 0, stream>>>(src, dst, ai, aj, (float*)mbuf, denom, alpha, EH, H1);
    self_agg_kernel<<<cdiv((long long)N * HC1 / 4, TPB), TPB, 0, stream>>>(
        (const float4*)xh, aself, (float*)mbuf, denom, (float4*)hbuf, N * HC1 / 4, C1 / 4);
    edge_agg_kernel<<<cdiv((long long)E * HC1 / 4, TPB), TPB, 0, stream>>>(
        src, dst, (const float4*)xh, alpha, hbuf, E * (HC1 / 4), H1, C1 / 4);
    post1_kernel<<<cdiv((long long)N * HC1 / 4, TPB), TPB, 0, stream>>>(
        (float4*)hbuf, bias1, feat16, N * HC1 / 4, HC1 / 4);

    // =========================== layer 2 ===========================
    cvt_f16_kernel<<<cdiv(HC2 * HC1, TPB), TPB, 0, stream>>>(lin2_w, w16, HC2 * HC1);
    {   // xh2 = h @ lin2_w^T : M=N, K=256, Ncols=128
        dim3 blk(32, 4), grd(N / 16, HC2 / 64);
        gemm_wmma_f16<<<grd, blk, 0, stream>>>(feat16, w16, xh, HC1, HC2);
    }
    NH = N * H2; EH = E * H2;
    logits_kernel<<<cdiv(NH, TPB), TPB, 0, stream>>>(xh, att2_i, att2_j, ai, aj, NH, H2, C2);
    init_self_kernel<<<cdiv(NH, TPB), TPB, 0, stream>>>(ai, aj, aself, mbuf, NH);
    edge_max_kernel<<<cdiv(EH, TPB), TPB, 0, stream>>>(src, dst, ai, aj, mbuf, EH, H2);
    denom_init_kernel<<<cdiv(NH, TPB), TPB, 0, stream>>>(aself, mbuf, denom, NH);
    edge_denom_kernel<<<cdiv(EH, TPB), TPB, 0, stream>>>(src, dst, ai, aj, (float*)mbuf, denom, EH, H2);
    edge_alpha_kernel<<<cdiv(EH, TPB), TPB, 0, stream>>>(src, dst, ai, aj, (float*)mbuf, denom, alpha, EH, H2);
    self_agg_kernel<<<cdiv((long long)N * HC2 / 4, TPB), TPB, 0, stream>>>(
        (const float4*)xh, aself, (float*)mbuf, denom, (float4*)out, N * HC2 / 4, C2 / 4);
    edge_agg_kernel<<<cdiv((long long)E * HC2 / 4, TPB), TPB, 0, stream>>>(
        src, dst, (const float4*)xh, alpha, out, E * (HC2 / 4), H2, C2 / 4);
    post2_kernel<<<cdiv((long long)N * HC2 / 4, TPB), TPB, 0, stream>>>(
        (float4*)out, bias2, N * HC2 / 4, HC2 / 4);
}
